// Mamba2MacOS_14302241096179
// MI455X (gfx1250) — matrix operand order repaired and checked by reading, not compiled, not run on hardware
//
#include <hip/hip_runtime.h>
#include <hip/hip_bf16.h>

// ---------------- problem constants ----------------
#define BATCH   2
#define SEQLEN  4096
#define DMODEL  1024
#define DINNER  2048
#define DSTATE  64
#define NHEADS  16
#define HEADDIM 128
#define NPROJ   (NHEADS + 2 * DSTATE)   // 144
#define MROWS   (BATCH * SEQLEN)        // 8192

// ---------------- WMMA vector types ----------------
typedef __attribute__((ext_vector_type(16))) __bf16 v16bf;
typedef __attribute__((ext_vector_type(8)))  float  v8f;

union FragBF {
    v16bf v;
    uint4 q[2];
};

__device__ __forceinline__ unsigned short f2bf(float f) {
    unsigned int x = __float_as_uint(f);
    unsigned int r = (x + 0x7FFFu + ((x >> 16) & 1u)) >> 16;   // RNE
    return (unsigned short)r;
}

// ---------------- fp32 -> bf16 cast ----------------
__global__ void k_cast_bf16(const float* __restrict__ src,
                            unsigned short* __restrict__ dst, int n) {
    int i = blockIdx.x * blockDim.x + threadIdx.x;
    if (i < n) dst[i] = f2bf(src[i]);
}

// ---------------- WMMA bf16 GEMM: C[M,N] = A[M,K] * B[N,K]^T (+bias) ----------------
// One wave computes a (16*MT) x (16*NT) tile. A/B fragments are double-buffered
// (ping-pong) so k+32 loads are in flight while WMMAs consume step k.
// Requires: M % (16*MT) == 0, N % (16*NT) == 0, K % 64 == 0.
template <int MT, int NT>
__global__ __launch_bounds__(256)
void k_gemm_bf16(const unsigned short* __restrict__ A,   // [M,K] row-major bf16
                 const unsigned short* __restrict__ Bm,  // [N,K] row-major bf16
                 const float* __restrict__ bias,         // [N] or nullptr
                 float* __restrict__ C,                   // [M,N] fp32
                 int M, int N, int K) {
    const int lane  = threadIdx.x & 31;
    const int wave  = threadIdx.x >> 5;
    const int gwave = blockIdx.x * (blockDim.x >> 5) + wave;
    const int tilesN = N / (16 * NT);
    const int tm = gwave / tilesN;            // (16*MT)-row tile index
    const int tn = (gwave % tilesN) * NT;     // base 16-col tile index
    if (tm * 16 * MT >= M) return;            // wave-uniform: EXEC stays all-ones

    const int m0    = tm * 16 * MT;
    const int arow  = m0 + (lane & 15);
    const int aKoff = (lane < 16) ? 0 : 8;    // A: lanes<16 hold K 0..7 & 16..23
    const int bcol0 = tn * 16 + (lane & 15);
    const int bKoff = (lane < 16) ? 0 : 8;    // B: lanes<16 hold K 0..7 & 16..23

    v8f acc[MT][NT];
#pragma unroll
    for (int i = 0; i < MT; ++i)
#pragma unroll
        for (int j = 0; j < NT; ++j)
            acc[i][j] = (v8f){0.f,0.f,0.f,0.f,0.f,0.f,0.f,0.f};

    FragBF a0[MT], b0[NT], a1[MT], b1[NT];

    auto loadA = [&](FragBF (&af)[MT], int k) {
#pragma unroll
        for (int i = 0; i < MT; ++i) {
            const unsigned short* ap = A + (size_t)(arow + 16 * i) * K + k + aKoff;
            af[i].q[0] = *(const uint4*)(ap);
            af[i].q[1] = *(const uint4*)(ap + 16);
            __builtin_prefetch((const void*)(ap + 512), 0, 0);   // global_prefetch_b8
        }
    };
    auto loadB = [&](FragBF (&bf)[NT], int k) {
#pragma unroll
        for (int j = 0; j < NT; ++j) {
            const unsigned short* bp = Bm + (size_t)(bcol0 + 16 * j) * K + k + bKoff;
            bf[j].q[0] = *(const uint4*)(bp);
            bf[j].q[1] = *(const uint4*)(bp + 16);
        }
    };
    auto mma = [&](FragBF (&af)[MT], FragBF (&bf)[NT]) {
#pragma unroll
        for (int i = 0; i < MT; ++i)
#pragma unroll
            for (int j = 0; j < NT; ++j)
                acc[i][j] = __builtin_amdgcn_wmma_f32_16x16x32_bf16(
                    false, af[i].v, false, bf[j].v, (short)0, acc[i][j], false, false);
    };

    // 2-stage software pipeline over K (K % 64 == 0 for all call sites)
    loadA(a0, 0);
    loadB(b0, 0);
    for (int k = 0; k < K; k += 64) {
        loadA(a1, k + 32);                 // next stage in flight ...
        loadB(b1, k + 32);
        mma(a0, b0);                       // ... while consuming current
        if (k + 64 < K) {
            loadA(a0, k + 64);
            loadB(b0, k + 64);
        }
        mma(a1, b1);
    }

    // C/D layout: element v of acc -> row m0 + 16*i + v + (lane>=16 ? 8 : 0)
    const int rbase = (lane < 16) ? 0 : 8;
    const int ccol  = lane & 15;
#pragma unroll
    for (int i = 0; i < MT; ++i) {
        const int crow = m0 + 16 * i + rbase;
#pragma unroll
        for (int j = 0; j < NT; ++j) {
            const int col = tn * 16 + 16 * j + ccol;
            const float bv = bias ? bias[col] : 0.f;
#pragma unroll
            for (int v = 0; v < 8; ++v)
                C[(size_t)(crow + v) * N + col] = acc[i][j][v] + bv;
        }
    }
}

// ---------------- depthwise conv(4) + SiLU, emits fp32 + bf16 ----------------
__global__ void k_conv_silu(const float* __restrict__ xz,        // [MROWS, 2*DINNER], x in cols [0,DINNER)
                            const float* __restrict__ conv_w,    // [DINNER,1,4]
                            const float* __restrict__ conv_b,    // [DINNER]
                            float* __restrict__ u,               // [MROWS, DINNER]
                            unsigned short* __restrict__ u_bf) { // [MROWS, DINNER]
    int idx = blockIdx.x * blockDim.x + threadIdx.x;
    if (idx >= MROWS * DINNER) return;
    const int c  = idx & (DINNER - 1);
    const int bl = idx >> 11;                  // DINNER = 2^11
    const int l  = bl & (SEQLEN - 1);
    const int b  = bl >> 12;                   // SEQLEN = 2^12
    float acc = conv_b[c];
#pragma unroll
    for (int j = 0; j < 4; ++j) {
        int lj = l - 3 + j;                    // left pad 3 (causal)
        float xv = (lj >= 0) ? xz[(size_t)(b * SEQLEN + lj) * (2 * DINNER) + c] : 0.f;
        acc += xv * conv_w[c * 4 + j];
    }
    float s = acc / (1.f + __expf(-acc));      // SiLU
    u[idx]    = s;
    u_bf[idx] = f2bf(s);
}

// ---------------- sequential SSM scan ----------------
// grid = BATCH * (DINNER/64) blocks, block = 64 threads.
// Each thread owns one channel d: 64-wide state + A row kept in registers.
__global__ __launch_bounds__(64)
void k_ssm_scan(const float* __restrict__ u,        // [MROWS, DINNER]
                const float* __restrict__ xproj,    // [MROWS, NPROJ]
                const float* __restrict__ dt_bias,  // [NHEADS]
                const float* __restrict__ A_log,    // [DINNER, DSTATE]
                float* __restrict__ ys) {           // [MROWS, DINNER]
    const int blk  = blockIdx.x;
    const int b    = blk >> 5;                 // DINNER/64 = 32 blocks per batch
    const int dblk = blk & 31;
    const int d    = dblk * 64 + threadIdx.x;  // channel owned by this thread
    const int h    = d >> 7;                   // head = d / HEADDIM (block-uniform)

    float Ar[DSTATE], st[DSTATE];
#pragma unroll
    for (int n = 0; n < DSTATE; ++n) {
        Ar[n] = -__expf(A_log[(size_t)d * DSTATE + n]);
        st[n] = 0.f;
    }
    const float dtb = dt_bias[h];
    const size_t rbase = (size_t)b * SEQLEN;

    for (int t = 0; t < SEQLEN; ++t) {
        const size_t row = rbase + t;
        const float* xp = xproj + row * NPROJ;
        float dtv = xp[h] + dtb;
        dtv = (dtv > 20.f) ? dtv : log1pf(__expf(dtv));   // softplus
        const float du = dtv * u[row * DINNER + d];
        const float* Bp = xp + NHEADS;                    // 64 B values (uniform bcast)
        const float* Cp = xp + NHEADS + DSTATE;           // 64 C values
        float y = 0.f;
#pragma unroll
        for (int n = 0; n < DSTATE; ++n) {
            float dA = __expf(dtv * Ar[n]);
            float s  = st[n] * dA + du * Bp[n];
            st[n] = s;
            y += s * Cp[n];
        }
        ys[row * DINNER + d] = y;
    }
}

// ---------------- y + u*D, gate by SiLU(z), LayerNorm, emit bf16 ----------------
__global__ __launch_bounds__(256)
void k_gate_norm(const float* __restrict__ ys,      // [MROWS, DINNER]
                 const float* __restrict__ u,       // [MROWS, DINNER]
                 const float* __restrict__ xz,      // z = cols [DINNER, 2*DINNER)
                 const float* __restrict__ Dv,      // [DINNER]
                 const float* __restrict__ norm_w,  // [DINNER]
                 const float* __restrict__ norm_b,  // [DINNER]
                 unsigned short* __restrict__ yn_bf) {
    const int row  = blockIdx.x;
    const int lane = threadIdx.x & 31;
    const int wv   = threadIdx.x >> 5;
    const float* zr = xz + (size_t)row * (2 * DINNER) + DINNER;
    __shared__ float s1[8], s2[8];

    float vals[8];
    float sum = 0.f, sumsq = 0.f;
#pragma unroll
    for (int i = 0; i < 8; ++i) {
        int dd = threadIdx.x + i * 256;
        float y = ys[(size_t)row * DINNER + dd] + u[(size_t)row * DINNER + dd] * Dv[dd];
        float z = zr[dd];
        float g = y * (z / (1.f + __expf(-z)));
        vals[i] = g;
        sum += g; sumsq += g * g;
    }
#pragma unroll
    for (int off = 16; off > 0; off >>= 1) {          // wave32 reduction
        sum   += __shfl_xor(sum,   off, 32);
        sumsq += __shfl_xor(sumsq, off, 32);
    }
    if (lane == 0) { s1[wv] = sum; s2[wv] = sumsq; }
    __syncthreads();
    if (threadIdx.x == 0) {
        float a = 0.f, q = 0.f;
        for (int i = 0; i < 8; ++i) { a += s1[i]; q += s2[i]; }
        float mu = a / (float)DINNER;
        s1[0] = mu;
        s2[0] = q / (float)DINNER - mu * mu;          // population variance
    }
    __syncthreads();
    const float mu  = s1[0];
    const float inv = rsqrtf(s2[0] + 1e-5f);
#pragma unroll
    for (int i = 0; i < 8; ++i) {
        int dd = threadIdx.x + i * 256;
        float g = (vals[i] - mu) * inv * norm_w[dd] + norm_b[dd];
        yn_bf[(size_t)row * DINNER + dd] = f2bf(g);
    }
}

// ---------------- host-side launch ----------------
extern "C" void kernel_launch(void* const* d_in, const int* in_sizes, int n_in,
                              void* d_out, int out_size, void* d_ws, size_t ws_size,
                              hipStream_t stream) {
    const float* hidden   = (const float*)d_in[0];   // [B,L,DMODEL]
    const float* in_w     = (const float*)d_in[1];   // [2*DINNER, DMODEL]
    const float* conv_w   = (const float*)d_in[2];   // [DINNER,1,4]
    const float* conv_b   = (const float*)d_in[3];   // [DINNER]
    const float* dt_w     = (const float*)d_in[4];   // [NPROJ, DINNER]
    const float* dt_b     = (const float*)d_in[5];   // [NPROJ]
    const float* dt_bias  = (const float*)d_in[6];   // [NHEADS]
    const float* A_log    = (const float*)d_in[7];   // [DINNER, DSTATE]
    const float* Dv       = (const float*)d_in[8];   // [DINNER]
    const float* norm_w   = (const float*)d_in[9];   // [DINNER]
    const float* norm_b   = (const float*)d_in[10];  // [DINNER]
    const float* out_w    = (const float*)d_in[11];  // [DMODEL, DINNER]
    float* out = (float*)d_out;                      // [B,L,DMODEL] fp32

    // bump allocator over d_ws (256B aligned)
    char* ws = (char*)d_ws;
    size_t off = 0;
    auto alloc = [&](size_t bytes) {
        void* p = ws + off;
        off += (bytes + 255) & ~(size_t)255;
        return p;
    };
    unsigned short* h_bf   = (unsigned short*)alloc((size_t)MROWS * DMODEL * 2);
    unsigned short* win_bf = (unsigned short*)alloc((size_t)2 * DINNER * DMODEL * 2);
    unsigned short* wdt_bf = (unsigned short*)alloc((size_t)NPROJ * DINNER * 2);
    unsigned short* wout_bf= (unsigned short*)alloc((size_t)DMODEL * DINNER * 2);
    float*          xz     = (float*)alloc((size_t)MROWS * 2 * DINNER * 4);
    float*          u      = (float*)alloc((size_t)MROWS * DINNER * 4);
    unsigned short* u_bf   = (unsigned short*)alloc((size_t)MROWS * DINNER * 2);
    float*          xproj  = (float*)alloc((size_t)MROWS * NPROJ * 4);
    float*          ys     = (float*)alloc((size_t)MROWS * DINNER * 4);
    unsigned short* yn_bf  = (unsigned short*)alloc((size_t)MROWS * DINNER * 2);
    (void)ws_size;

    // 1) bf16 casts
    {
        int n;
        n = MROWS * DMODEL;
        k_cast_bf16<<<(n + 255) / 256, 256, 0, stream>>>(hidden, h_bf, n);
        n = 2 * DINNER * DMODEL;
        k_cast_bf16<<<(n + 255) / 256, 256, 0, stream>>>(in_w, win_bf, n);
        n = NPROJ * DINNER;
        k_cast_bf16<<<(n + 255) / 256, 256, 0, stream>>>(dt_w, wdt_bf, n);
        n = DMODEL * DINNER;
        k_cast_bf16<<<(n + 255) / 256, 256, 0, stream>>>(out_w, wout_bf, n);
    }

    // 2) in_proj GEMM: xz[8192,4096] = h[8192,1024] * in_w^T   (wave tile 32x64)
    {
        int waves = (MROWS / 32) * ((2 * DINNER) / 64);
        k_gemm_bf16<2, 4><<<(waves + 7) / 8, 256, 0, stream>>>(
            h_bf, win_bf, nullptr, xz, MROWS, 2 * DINNER, DMODEL);
    }

    // 3) causal depthwise conv + SiLU
    {
        int n = MROWS * DINNER;
        k_conv_silu<<<(n + 255) / 256, 256, 0, stream>>>(xz, conv_w, conv_b, u, u_bf);
    }

    // 4) dt/B/C projection GEMM: xproj[8192,144] = u * dt_w^T + dt_b   (N=144 -> 16x16 tiles)
    {
        int waves = (MROWS / 16) * (NPROJ / 16);
        k_gemm_bf16<1, 1><<<(waves + 7) / 8, 256, 0, stream>>>(
            u_bf, wdt_bf, dt_b, xproj, MROWS, NPROJ, DINNER);
    }

    // 5) sequential SSM scan
    k_ssm_scan<<<BATCH * (DINNER / 64), 64, 0, stream>>>(u, xproj, dt_bias, A_log, ys);

    // 6) gate + layernorm -> bf16
    k_gate_norm<<<MROWS, 256, 0, stream>>>(ys, u, xz, Dv, norm_w, norm_b, yn_bf);

    // 7) out_proj GEMM: out[8192,1024] = yn * out_w^T   (wave tile 32x64)
    {
        int waves = (MROWS / 32) * (DMODEL / 64);
        k_gemm_bf16<2, 4><<<(waves + 7) / 8, 256, 0, stream>>>(
            yn_bf, wout_bf, nullptr, out, MROWS, DMODEL, DINNER);
    }
    (void)in_sizes; (void)n_in; (void)out_size;
}
